// RoPECrossAttention_5351529251428
// MI455X (gfx1250) — matrix operand order repaired and checked by reading, not compiled, not run on hardware
//
#include <hip/hip_runtime.h>
#include <hip/hip_bf16.h>
#include <cstdint>
#include <cstddef>

// ---------------------------------------------------------------------------
// RoPE cross-attention for MI455X (gfx1250), bf16 WMMA pipeline with
// double-buffered async global->LDS staging (ASYNCcnt pipelined).
// B=8, NQ=1024, NKV=2048, D=1024, H=16, HD=64.
// ---------------------------------------------------------------------------

#ifndef USE_ASYNC_COPY
#define USE_ASYNC_COPY 1   // global_load_async_to_lds_b128 path (CDNA5)
#endif

#define B_   8
#define NQ_  1024
#define NKV_ 2048
#define D_   1024
#define H_   16
#define HD_  64

typedef __attribute__((ext_vector_type(16))) __bf16 bf16x16;
typedef __attribute__((ext_vector_type(8)))  __bf16 bf16x8;
typedef __attribute__((ext_vector_type(8)))  float  floatx8;

// ---------------- scalar helpers ----------------
static __device__ __forceinline__ unsigned short f2bfbits(float f) {
    union { float f; unsigned u; } c; c.f = f;
    unsigned r = c.u + 0x7FFFu + ((c.u >> 16) & 1u);   // round-to-nearest-even
    return (unsigned short)(r >> 16);
}
static __device__ __forceinline__ float bfbits2f(unsigned short u) {
    union { unsigned u; float f; } c; c.u = ((unsigned)u) << 16;
    return c.f;
}
static __device__ __forceinline__ unsigned ldsaddr(const void* p) {
    // For LDS objects the low 32 bits of the generic address are the LDS offset.
    return (unsigned)(size_t)p;
}

// ---------------- async global->LDS copy (16 bytes per lane) ----------------
static __device__ __forceinline__ void copy16B(unsigned lds_a,
                                               const unsigned short* g,
                                               unsigned short* l) {
#if USE_ASYNC_COPY
    (void)l;
    asm volatile("global_load_async_to_lds_b128 %0, %1, off"
                 :: "v"(lds_a), "v"((unsigned long long)(size_t)g)
                 : "memory");
#else
    (void)lds_a;
    *(uint4*)l = *(const uint4*)g;
#endif
}
// Wait until at most N async copies remain outstanding (in-order completion).
template <int N>
static __device__ __forceinline__ void wait_async_le() {
#if USE_ASYNC_COPY
    asm volatile("s_wait_asynccnt %0" :: "i"(N) : "memory");
#endif
}

// ---------------- WMMA wrapper ----------------
static __device__ __forceinline__ floatx8 wmma_bf16(bf16x16 a, bf16x16 b, floatx8 c) {
    // D = A(16x32 bf16) x B(32x16 bf16) + C(16x16 f32)
    return __builtin_amdgcn_wmma_f32_16x16x32_bf16(
        /*neg_a=*/false, a, /*neg_b=*/false, b,
        /*c_mod=*/(short)0, c, /*reuse_a=*/false, /*reuse_b=*/false);
}

// ---------------- fragment builders (ISA §7.12.2 layouts) ----------------
// A matrix 16x32 bf16, lane L -> row M=L&15, baseK = (L>=16)*8.
// Fragment elements are two contiguous 16B chunks: K=[baseK..baseK+7] and
// K=[baseK+16..baseK+23]  -> two b128 loads, no scalar packing.
static __device__ __forceinline__ bf16x16 load_frag_a(const unsigned short* base,
                                                      int ld, int m0, int k0, int lane) {
    const unsigned short* p = base + (size_t)(m0 + (lane & 15)) * ld
                                   + k0 + ((lane >> 4) << 3);
    bf16x8 lo = *(const bf16x8*)p;         // K = baseK .. baseK+7
    bf16x8 hi = *(const bf16x8*)(p + 16);  // K = baseK+16 .. baseK+23
    return __builtin_shufflevector(lo, hi, 0, 1, 2, 3, 4, 5, 6, 7,
                                           8, 9, 10, 11, 12, 13, 14, 15);
}
// B matrix 32x16 bf16 from a tile stored transposed [N][K]:
// lane L -> col N=L&15, baseK = (L>=16)*8; elements are K=[baseK..baseK+7]
// and K=[baseK+16..baseK+23] -> two b128 loads.
static __device__ __forceinline__ bf16x16 load_frag_bt(const unsigned short* base,
                                                       int ld, int n0, int k0, int lane) {
    const unsigned short* p = base + (size_t)(n0 + (lane & 15)) * ld
                                   + k0 + ((lane >> 4) << 3);
    bf16x8 lo = *(const bf16x8*)p;
    bf16x8 hi = *(const bf16x8*)(p + 16);
    return __builtin_shufflevector(lo, hi, 0, 1, 2, 3, 4, 5, 6, 7,
                                           8, 9, 10, 11, 12, 13, 14, 15);
}

// ===========================================================================
// Kernel 1: fused LayerNorm (over D=1024) + cast fp32 -> bf16
// ===========================================================================
__global__ __launch_bounds__(256) void ln_cast_kernel(const float* __restrict__ x,
                                                      unsigned short* __restrict__ y) {
    const int row = blockIdx.x;
    const int t = threadIdx.x;
    const float* xr = x + (size_t)row * D_;
    float lv[4], s = 0.f, s2 = 0.f;
    const int i0 = t * 4;
#pragma unroll
    for (int j = 0; j < 4; ++j) { float v = xr[i0 + j]; lv[j] = v; s += v; s2 += v * v; }
    __shared__ float rs[256], rs2[256];
    rs[t] = s; rs2[t] = s2;
    __syncthreads();
    for (int off = 128; off > 0; off >>= 1) {
        if (t < off) { rs[t] += rs[t + off]; rs2[t] += rs2[t + off]; }
        __syncthreads();
    }
    const float mu  = rs[0] * (1.0f / D_);
    const float var = rs2[0] * (1.0f / D_) - mu * mu;
    const float inv = rsqrtf(var + 1e-5f);
    unsigned short* yr = y + (size_t)row * D_;
#pragma unroll
    for (int j = 0; j < 4; ++j) yr[i0 + j] = f2bfbits((lv[j] - mu) * inv);
}

// ===========================================================================
// Kernel 2: fp32 -> bf16 weight cast (grid-stride)
// ===========================================================================
__global__ __launch_bounds__(256) void cast_f2bf_kernel(const float* __restrict__ x,
                                                        unsigned short* __restrict__ y, int n) {
    for (int i = blockIdx.x * 256 + threadIdx.x; i < n; i += gridDim.x * 256)
        y[i] = f2bfbits(x[i]);
}

// ===========================================================================
// Kernel 3: tiled bf16 GEMM  C[M,N] = A[M,K] * W[N,K]^T + bias
//   128x128 block tile, 8 waves, wave = 32x64 (2x4 WMMA accums), K step 32.
//   Double-buffered LDS with async prefetch of tile i+1 during compute of i.
//   OUT_MODE: 0 = bf16 row-major, 1 = f32 row-major, 2 = bf16 transposed
//   per-batch (Vt layout [B, D, NKV], rows are b*NKV+key).
// ===========================================================================
template <int OUT_MODE>
__global__ __launch_bounds__(256) void gemm_bias_kernel(
    const unsigned short* __restrict__ A, const unsigned short* __restrict__ W,
    const float* __restrict__ bias, unsigned short* __restrict__ outB,
    float* __restrict__ outF, int M, int N, int K) {
    __shared__ unsigned short As[2][128][40];  // 80B row stride (16B aligned)
    __shared__ unsigned short Ws[2][128][40];

    const int t = threadIdx.x;
    const int lane = t & 31, wid = t >> 5;
    const int bm = blockIdx.x * 128, bn = blockIdx.y * 128;
    const int wm = (wid & 3) * 32, wn = (wid >> 2) * 64;

    // cooperative copy mapping: thread -> (row, 32B segment)
    const int crow = t >> 1;
    const int cseg = (t & 1) << 4;  // ushort offset 0 or 16
    const unsigned short* gA = A + (size_t)(bm + crow) * K + cseg;
    const unsigned short* gW = W + (size_t)(bn + crow) * K + cseg;

    auto issue_tile = [&](int k0, int buf) {
        unsigned short* lA = &As[buf][crow][cseg];
        unsigned short* lW = &Ws[buf][crow][cseg];
        copy16B(ldsaddr(lA),     gA + k0,     lA);
        copy16B(ldsaddr(lA) + 16, gA + k0 + 8, lA + 8);
        copy16B(ldsaddr(lW),     gW + k0,     lW);
        copy16B(ldsaddr(lW) + 16, gW + k0 + 8, lW + 8);
    };

    const floatx8 zero8 = {0.f, 0.f, 0.f, 0.f, 0.f, 0.f, 0.f, 0.f};
    floatx8 acc[2][4];
#pragma unroll
    for (int i = 0; i < 2; ++i)
#pragma unroll
        for (int j = 0; j < 4; ++j) acc[i][j] = zero8;

    const int NT = K / 32;
    issue_tile(0, 0);
    for (int i = 0; i < NT; ++i) {
        const int buf = i & 1;
        if (i + 1 < NT) {
            // prefetch next tile into the other buffer (freed by end-of-prev-
            // iteration barrier); 4 new async ops per wave go in flight.
            issue_tile((i + 1) * 32, buf ^ 1);
            wait_async_le<4>();   // tile i's 4 copies complete (in-order)
        } else {
            wait_async_le<0>();
        }
        __syncthreads();

        bf16x16 a0 = load_frag_a(&As[buf][0][0], 40, wm,      0, lane);
        bf16x16 a1 = load_frag_a(&As[buf][0][0], 40, wm + 16, 0, lane);
#pragma unroll
        for (int nt = 0; nt < 4; ++nt) {
            bf16x16 bfr = load_frag_bt(&Ws[buf][0][0], 40, wn + nt * 16, 0, lane);
            acc[0][nt] = wmma_bf16(a0, bfr, acc[0][nt]);
            acc[1][nt] = wmma_bf16(a1, bfr, acc[1][nt]);
        }
        __syncthreads();   // compute done before this buffer is refilled
    }

    const int hi = lane >> 4, ln = lane & 15;
#pragma unroll
    for (int mt = 0; mt < 2; ++mt)
#pragma unroll
        for (int nt = 0; nt < 4; ++nt)
#pragma unroll
            for (int r = 0; r < 8; ++r) {
                const int row = bm + wm + mt * 16 + r + 8 * hi;
                const int col = bn + wn + nt * 16 + ln;
                const float v = acc[mt][nt][r] + bias[col];
                if (OUT_MODE == 1) {
                    outF[(size_t)row * N + col] = v;
                } else if (OUT_MODE == 2) {
                    const int bb  = row >> 11;      // row / NKV_
                    const int key = row & (NKV_ - 1);
                    outB[((size_t)(bb * D_ + col)) * NKV_ + key] = f2bfbits(v);
                } else {
                    outB[(size_t)row * N + col] = f2bfbits(v);
                }
            }
}

// ===========================================================================
// Kernel 4: in-place RoPE on bf16 [rows, D]; cos/sin fp32 [npos, HD]
// ===========================================================================
__global__ __launch_bounds__(256) void rope_kernel(unsigned short* __restrict__ x,
                                                   const float* __restrict__ cosb,
                                                   const float* __restrict__ sinb,
                                                   int nrows, int npos) {
    const int total = nrows * (D_ / 2);
    const int id = blockIdx.x * 256 + threadIdx.x;
    if (id >= total) return;
    const int row = id / (D_ / 2);
    const int c   = id % (D_ / 2);
    const int h   = c >> 5;          // head
    const int d   = c & 31;          // first half dim
    const int pos = row % npos;
    unsigned short* p = x + (size_t)row * D_ + h * HD_;
    const float x1 = bfbits2f(p[d]);
    const float x2 = bfbits2f(p[d + 32]);
    const float* cr = cosb + (size_t)pos * HD_;
    const float* sr = sinb + (size_t)pos * HD_;
    p[d]      = f2bfbits(x1 * cr[d]      - x2 * sr[d]);
    p[d + 32] = f2bfbits(x2 * cr[d + 32] + x1 * sr[d + 32]);
}

// ===========================================================================
// Kernel 5: flash attention.  Block = (b, h, 128 q rows), 8 waves x 16 rows.
//   K tiles key-major, V tiles d-major (from Vt) -> all fragments are
//   contiguous b128 LDS loads.  Double-buffered async K/V staging.
// ===========================================================================
__global__ __launch_bounds__(256) void attn_kernel(const unsigned short* __restrict__ Qb,
                                                   const unsigned short* __restrict__ Kb,
                                                   const unsigned short* __restrict__ Vt,
                                                   unsigned short* __restrict__ ctx) {
    const int b = blockIdx.z, h = blockIdx.y, q0 = blockIdx.x * 128;
    const int t = threadIdx.x;
    const int lane = t & 31, wid = t >> 5;
    const int hi = lane >> 4, ln = lane & 15;

    __shared__ unsigned short Ks[2][64][72];   // [key][d], 2 x 9.2 KB
    __shared__ unsigned short Vs[2][64][72];   // [d][key], 2 x 9.2 KB
    __shared__ unsigned short Ps[8][16][72];   // per-wave P tile, 18.4 KB

    // Q fragments for this wave's 16 rows (kept in registers for entire pass)
    const unsigned short* Qbase =
        Qb + (size_t)(b * NQ_ + q0 + wid * 16) * D_ + h * HD_;
    const bf16x16 qa0 = load_frag_a(Qbase, D_, 0, 0,  lane);
    const bf16x16 qa1 = load_frag_a(Qbase, D_, 0, 32, lane);

    const floatx8 zero8 = {0.f, 0.f, 0.f, 0.f, 0.f, 0.f, 0.f, 0.f};
    floatx8 acc[4];
#pragma unroll
    for (int nt = 0; nt < 4; ++nt) acc[nt] = zero8;
    float mrun[8], lrun[8];
#pragma unroll
    for (int r = 0; r < 8; ++r) { mrun[r] = -3.0e38f; lrun[r] = 0.f; }
    const float scale = 0.125f;  // 1/sqrt(HD)

    // cooperative tile copy mapping: thread -> (row, 32B segment)
    const int kr  = t >> 2;        // K: key row / V: d row
    const int seg = (t & 3) << 4;  // ushort offset: 0,16,32,48
    const unsigned short* gK = Kb + (size_t)(b * NKV_ + kr) * D_ + h * HD_ + seg;
    const unsigned short* gV = Vt + (size_t)(b * D_ + h * HD_ + kr) * NKV_ + seg;

    auto issue_tile = [&](int kv0, int buf) {
        unsigned short* lK = &Ks[buf][kr][seg];
        unsigned short* lV = &Vs[buf][kr][seg];
        const unsigned short* gk = gK + (size_t)kv0 * D_;  // advance keys (rows)
        const unsigned short* gv = gV + kv0;               // advance keys (cols)
        copy16B(ldsaddr(lK),     gk,     lK);
        copy16B(ldsaddr(lK) + 16, gk + 8, lK + 8);
        copy16B(ldsaddr(lV),     gv,     lV);
        copy16B(ldsaddr(lV) + 16, gv + 8, lV + 8);
    };

    const int NT = NKV_ / 64;
    issue_tile(0, 0);
    for (int i = 0; i < NT; ++i) {
        const int buf = i & 1;
        if (i + 1 < NT) { issue_tile((i + 1) * 64, buf ^ 1); wait_async_le<4>(); }
        else            { wait_async_le<0>(); }
        __syncthreads();

        // S = Q * K^T  (16 x 64)
        floatx8 s[4];
#pragma unroll
        for (int nt = 0; nt < 4; ++nt) {
            s[nt] = zero8;
            bf16x16 b0 = load_frag_bt(&Ks[buf][0][0], 72, nt * 16, 0,  lane);
            s[nt] = wmma_bf16(qa0, b0, s[nt]);
            bf16x16 b1 = load_frag_bt(&Ks[buf][0][0], 72, nt * 16, 32, lane);
            s[nt] = wmma_bf16(qa1, b1, s[nt]);
        }

        // online softmax in WMMA C layout; row stats reduced across the
        // 16-lane N group with shuffles.
#pragma unroll
        for (int r = 0; r < 8; ++r) {
            float mx = fmaxf(fmaxf(s[0][r], s[1][r]), fmaxf(s[2][r], s[3][r])) * scale;
#pragma unroll
            for (int off = 1; off < 16; off <<= 1) mx = fmaxf(mx, __shfl_xor(mx, off, 32));
            const float mnew = fmaxf(mrun[r], mx);
            const float corr = __expf(mrun[r] - mnew);
            float sum = 0.f;
#pragma unroll
            for (int nt = 0; nt < 4; ++nt) {
                const float p = __expf(s[nt][r] * scale - mnew);
                sum += p;
                Ps[wid][r + 8 * hi][nt * 16 + ln] = f2bfbits(p);
                acc[nt][r] *= corr;
            }
#pragma unroll
            for (int off = 1; off < 16; off <<= 1) sum += __shfl_xor(sum, off, 32);
            lrun[r] = lrun[r] * corr + sum;
            mrun[r] = mnew;
        }

        // ctx += P * V  (P via per-wave LDS; V tile is d-major so B-frags are
        // contiguous b128 reads; per-wave DS ordering makes Ps safe w/o barrier)
#pragma unroll
        for (int ks = 0; ks < 2; ++ks) {
            bf16x16 pa = load_frag_a(&Ps[wid][0][0], 72, 0, ks * 32, lane);
#pragma unroll
            for (int nt = 0; nt < 4; ++nt) {
                bf16x16 vb = load_frag_bt(&Vs[buf][0][0], 72, nt * 16, ks * 32, lane);
                acc[nt] = wmma_bf16(pa, vb, acc[nt]);
            }
        }
        __syncthreads();  // this buffer free for the prefetch 2 tiles ahead
    }

    // normalize and write bf16 context [B, NQ, D] at head column block
    const size_t rbase = (size_t)(b * NQ_ + q0 + wid * 16);
#pragma unroll
    for (int nt = 0; nt < 4; ++nt)
#pragma unroll
        for (int r = 0; r < 8; ++r) {
            const int m = r + 8 * hi;
            const float v = acc[nt][r] / lrun[r];
            ctx[(rbase + m) * D_ + h * HD_ + nt * 16 + ln] = f2bfbits(v);
        }
}

// ===========================================================================
// Host side
// ===========================================================================
extern "C" void kernel_launch(void* const* d_in, const int* in_sizes, int n_in,
                              void* d_out, int out_size, void* d_ws, size_t ws_size,
                              hipStream_t stream) {
    (void)in_sizes; (void)n_in; (void)out_size; (void)ws_size;

    const float* q_in   = (const float*)d_in[0];
    const float* kv_in  = (const float*)d_in[1];
    const float* q_cos  = (const float*)d_in[2];
    const float* q_sin  = (const float*)d_in[3];
    const float* kv_cos = (const float*)d_in[4];
    const float* kv_sin = (const float*)d_in[5];
    const float* Wq = (const float*)d_in[6];
    const float* bq = (const float*)d_in[7];
    const float* Wk = (const float*)d_in[8];
    const float* bk = (const float*)d_in[9];
    const float* Wv = (const float*)d_in[10];
    const float* bv = (const float*)d_in[11];
    const float* Wo = (const float*)d_in[12];
    const float* bo = (const float*)d_in[13];
    float* out = (float*)d_out;

    // workspace layout (bf16 buffers), ~160 MB total
    constexpr size_t SZ_QN  = (size_t)B_ * NQ_ * D_ * 2;   // 16 MiB
    constexpr size_t SZ_KVN = (size_t)B_ * NKV_ * D_ * 2;  // 32 MiB
    constexpr size_t SZ_W   = (size_t)D_ * D_ * 2;         //  2 MiB
    char* ws = (char*)d_ws;
    unsigned short* qn  = (unsigned short*)(ws);
    unsigned short* kvn = (unsigned short*)(ws + SZ_QN);
    unsigned short* wq  = (unsigned short*)(ws + SZ_QN + SZ_KVN);
    unsigned short* wk  = (unsigned short*)(ws + SZ_QN + SZ_KVN + SZ_W);
    unsigned short* wv  = (unsigned short*)(ws + SZ_QN + SZ_KVN + 2 * SZ_W);
    unsigned short* wo  = (unsigned short*)(ws + SZ_QN + SZ_KVN + 3 * SZ_W);
    unsigned short* Qb  = (unsigned short*)(ws + SZ_QN + SZ_KVN + 4 * SZ_W);
    unsigned short* Kb  = (unsigned short*)(ws + 2 * SZ_QN + SZ_KVN + 4 * SZ_W);
    unsigned short* Vt  = (unsigned short*)(ws + 2 * SZ_QN + 2 * SZ_KVN + 4 * SZ_W);
    unsigned short* ctx = (unsigned short*)(ws + 2 * SZ_QN + 3 * SZ_KVN + 4 * SZ_W);

    // 1) LayerNorm + cast
    ln_cast_kernel<<<B_ * NQ_,  256, 0, stream>>>(q_in,  qn);
    ln_cast_kernel<<<B_ * NKV_, 256, 0, stream>>>(kv_in, kvn);

    // 2) weight casts
    cast_f2bf_kernel<<<512, 256, 0, stream>>>(Wq, wq, D_ * D_);
    cast_f2bf_kernel<<<512, 256, 0, stream>>>(Wk, wk, D_ * D_);
    cast_f2bf_kernel<<<512, 256, 0, stream>>>(Wv, wv, D_ * D_);
    cast_f2bf_kernel<<<512, 256, 0, stream>>>(Wo, wo, D_ * D_);

    // 3) projections (bf16 out; V written transposed per batch -> Vt[B,D,NKV])
    gemm_bias_kernel<0><<<dim3(B_ * NQ_ / 128,  D_ / 128), 256, 0, stream>>>(
        qn,  wq, bq, Qb, nullptr, B_ * NQ_,  D_, D_);
    gemm_bias_kernel<0><<<dim3(B_ * NKV_ / 128, D_ / 128), 256, 0, stream>>>(
        kvn, wk, bk, Kb, nullptr, B_ * NKV_, D_, D_);
    gemm_bias_kernel<2><<<dim3(B_ * NKV_ / 128, D_ / 128), 256, 0, stream>>>(
        kvn, wv, bv, Vt, nullptr, B_ * NKV_, D_, D_);

    // 4) RoPE (in place on bf16 Q and K)
    rope_kernel<<<(B_ * NQ_  * (D_ / 2)) / 256, 256, 0, stream>>>(Qb, q_cos,  q_sin,  B_ * NQ_,  NQ_);
    rope_kernel<<<(B_ * NKV_ * (D_ / 2)) / 256, 256, 0, stream>>>(Kb, kv_cos, kv_sin, B_ * NKV_, NKV_);

    // 5) flash attention -> bf16 context
    attn_kernel<<<dim3(NQ_ / 128, H_, B_), 256, 0, stream>>>(Qb, Kb, Vt, ctx);

    // 6) output projection (fp32 out + bias)
    gemm_bias_kernel<1><<<dim3(B_ * NQ_ / 128, D_ / 128), 256, 0, stream>>>(
        ctx, wo, bo, nullptr, out, B_ * NQ_, D_, D_);
}
